// get_model_16372415332415
// MI455X (gfx1250) — matrix operand order repaired and checked by reading, not compiled, not run on hardware
//
#include <hip/hip_runtime.h>
#include <hip/hip_bf16.h>
#include <cstdint>
#include <cstddef>

typedef __attribute__((ext_vector_type(16))) _Float16 v16h;
typedef __attribute__((ext_vector_type(8)))  float    v8f;

// ---------------------------------------------------------------- constants
#define BB   32
#define NPT  2048
#define S1   512
#define NS1  32
#define S2   128
#define NS2  64

// ---------------------------------------------------------------- WMMA utils
__device__ __forceinline__ v8f wmma16(v16h a, v16h b, v8f c) {
  return __builtin_amdgcn_wmma_f32_16x16x32_f16(
      /*neg_a=*/false, a, /*neg_b=*/false, b,
      /*c_mod=*/(short)0, c, /*reuse_a=*/false, /*reuse_b=*/false);
}

// A fragment: 16x32 f16, row-major source with leading dim ld.
// Layout: lanes0-15 M=0..15; element h -> K = h (lanes0-15) / 16+h (lanes16-31).
__device__ __forceinline__ v16h load_a16(const _Float16* X, int ld, int m0, int k0, int lane) {
  union { v16h v; _Float16 e[16]; } A;
  int m = m0 + (lane & 15);
#pragma unroll
  for (int h = 0; h < 16; ++h) {
    int kk = ((lane < 16) ? 0 : 16) + h;
    A.e[h] = X[(size_t)m * ld + k0 + kk];
  }
  return A.v;
}

// B fragment from pre-packed weights: one aligned 32B load per lane.
// P layout: [kt][nt][lane][16 halves], NT = N/16.
__device__ __forceinline__ v16h load_b16p(const _Float16* P, int NT, int kt, int nt, int lane) {
  return *(const v16h*)(P + (((size_t)kt * NT + nt) * 32 + lane) * 16);
}

// D 16x16 f32: vgpr r -> M = r (lanes0-15) / r+8 (lanes16-31), N = lane&15.
// Applies y = relu(g*(d + bias) + bt), stores f16.
__device__ __forceinline__ void store_d_bnrelu(_Float16* Y, int ld, int m0, int n0, int lane,
                                               v8f d, const float* bias, const float* g,
                                               const float* bt) {
  union { v8f v; float e[8]; } D; D.v = d;
  int n = n0 + (lane & 15);
  float gg = g[n], bb = bias[n], tt = bt[n];
#pragma unroll
  for (int r = 0; r < 8; ++r) {
    int m = m0 + r + ((lane >= 16) ? 8 : 0);
    float y = gg * (D.e[r] + bb) + tt;
    Y[(size_t)m * ld + n] = (_Float16)fmaxf(y, 0.0f);
  }
}

// ---------------------------------------------------------------- prep kernels
__global__ void transpose_xyz_kernel(const float* __restrict__ xyz, float* __restrict__ pts) {
  int t = blockIdx.x * blockDim.x + threadIdx.x;           // BB*NPT*3
  if (t >= BB * NPT * 3) return;
  int b = t / (NPT * 3);
  int rem = t - b * NPT * 3;
  int n = rem / 3, c = rem - n * 3;
  pts[t] = xyz[(size_t)b * 3 * NPT + (size_t)c * NPT + n];
}

// Pack KxN f32 weights into WMMA-B fragment order (f16).
// out[((kt*NT+nt)*32+lane)*16+h] = W[(kt*32+((lane<16)?0:16)+h)*N + nt*16+(lane&15)]
__global__ void pack_wmma_b_kernel(const float* __restrict__ W, _Float16* __restrict__ P,
                                   int K, int N) {
  int t = blockIdx.x * blockDim.x + threadIdx.x;
  if (t >= K * N) return;
  int NT = N >> 4;
  int h = t & 15;
  int lane = (t >> 4) & 31;
  int tile = t >> 9;
  int nt = tile % NT;
  int kt = tile / NT;
  int k = kt * 32 + ((lane < 16) ? 0 : 16) + h;
  int n = nt * 16 + (lane & 15);
  P[t] = (_Float16)W[(size_t)k * N + n];
}

// ---------------------------------------------------------------- FPS
__global__ __launch_bounds__(256) void fps_kernel(const float* __restrict__ pts,
                                                  int* __restrict__ outIdx, int N, int S) {
  int b = blockIdx.x;
  __shared__ float dist[NPT];
  __shared__ float sval[256];
  __shared__ int   sidx[256];
  __shared__ int   sfar;
  const float* P = pts + (size_t)b * N * 3;
  for (int i = threadIdx.x; i < N; i += 256) dist[i] = 1e10f;
  if (threadIdx.x == 0) sfar = 0;
  __syncthreads();
  for (int it = 0; it < S; ++it) {
    int far = sfar;
    if (threadIdx.x == 0) outIdx[b * S + it] = far;
    float fx = P[far * 3 + 0], fy = P[far * 3 + 1], fz = P[far * 3 + 2];
    float best = -1.0f; int bi = 0;
    for (int i = threadIdx.x; i < N; i += 256) {
      float dx = P[i * 3 + 0] - fx, dy = P[i * 3 + 1] - fy, dz = P[i * 3 + 2] - fz;
      float d = dx * dx + dy * dy + dz * dz;
      float dm = fminf(dist[i], d);
      dist[i] = dm;
      if (dm > best) { best = dm; bi = i; }
    }
    sval[threadIdx.x] = best; sidx[threadIdx.x] = bi;
    __syncthreads();
    for (int off = 128; off > 0; off >>= 1) {
      if (threadIdx.x < off) {
        float v2 = sval[threadIdx.x + off]; int i2 = sidx[threadIdx.x + off];
        if (v2 > sval[threadIdx.x] || (v2 == sval[threadIdx.x] && i2 < sidx[threadIdx.x])) {
          sval[threadIdx.x] = v2; sidx[threadIdx.x] = i2;
        }
      }
      __syncthreads();
    }
    if (threadIdx.x == 0) sfar = sidx[0];
    __syncthreads();
  }
}

__global__ void gather_xyz_kernel(const float* __restrict__ pts, const int* __restrict__ fidx,
                                  float* __restrict__ out, int N, int S, int total) {
  int t = blockIdx.x * blockDim.x + threadIdx.x;           // BB*S
  if (t >= total) return;
  int b = t / S;
  int id = fidx[t];
  out[(size_t)t * 3 + 0] = pts[((size_t)b * N + id) * 3 + 0];
  out[(size_t)t * 3 + 1] = pts[((size_t)b * N + id) * 3 + 1];
  out[(size_t)t * 3 + 2] = pts[((size_t)b * N + id) * 3 + 2];
}

// ---------------------------------------------------------------- ball query (1 wave / centroid)
__global__ __launch_bounds__(32) void ballquery_kernel(const float* __restrict__ pts,
                                                       const float* __restrict__ cent,
                                                       int* __restrict__ gidx,
                                                       int N, int S, int nsample, float r2) {
  int bs = blockIdx.x;
  int b = bs / S;
  int lane = threadIdx.x;
  const float* P = pts + (size_t)b * N * 3;
  float cx = cent[(size_t)bs * 3 + 0], cy = cent[(size_t)bs * 3 + 1], cz = cent[(size_t)bs * 3 + 2];
  int* out = gidx + (size_t)bs * nsample;
  int count = 0, first = 0;
  for (int base = 0; base < N; base += 32) {
    int p = base + lane;
    float dx = P[p * 3 + 0] - cx, dy = P[p * 3 + 1] - cy, dz = P[p * 3 + 2] - cz;
    bool in = (dx * dx + dy * dy + dz * dz) <= r2;
    unsigned mask = (unsigned)__ballot(in);
    if (count == 0 && mask) first = base + __ffs((int)mask) - 1;
    if (in) {
      int pos = count + __popc(mask & ((1u << lane) - 1u));
      if (pos < nsample) out[pos] = p;
    }
    count += __popc(mask);
    if (count >= nsample) break;
  }
  if (count > nsample) count = nsample;
  for (int i = count + lane; i < nsample; i += 32) out[i] = first;
}

// ---------------------------------------------------------------- SA1 fused: rows=32, 3->64->64->128, max
__global__ __launch_bounds__(128) void sa1_fused_kernel(
    const float* __restrict__ pts, const float* __restrict__ cent, const int* __restrict__ gidx,
    const float* W1, const float* b1, const float* g1, const float* t1,
    const _Float16* W2p, const float* b2, const float* g2, const float* t2,
    const _Float16* W3p, const float* b3, const float* g3, const float* t3,
    float* __restrict__ outp) {
  __shared__ _Float16 h1[32 * 64];
  __shared__ _Float16 h2[32 * 64];
  __shared__ _Float16 h3[32 * 128];
  const int bs = blockIdx.x;               // b*512 + s
  const int b = bs >> 9;
  const int tid = threadIdx.x, wave = tid >> 5, lane = tid & 31;

  const float cx = cent[(size_t)bs * 3 + 0], cy = cent[(size_t)bs * 3 + 1], cz = cent[(size_t)bs * 3 + 2];
  {  // layer0: 3 -> 64 (scalar)
    int r = tid >> 2;
    int c0 = (tid & 3) * 16;
    int id = gidx[(size_t)bs * NS1 + r];
    const float* P = pts + ((size_t)b * NPT + id) * 3;
    float px = P[0] - cx, py = P[1] - cy, pz = P[2] - cz;
    for (int c = c0; c < c0 + 16; ++c) {
      float acc = px * W1[c] + py * W1[64 + c] + pz * W1[128 + c] + b1[c];
      float y = g1[c] * acc + t1[c];
      h1[r * 64 + c] = (_Float16)fmaxf(y, 0.0f);
    }
  }
  __syncthreads();
  {  // layer1: (32x64)@(64x64) WMMA, NT=4
    int ni = wave;
    for (int mi = 0; mi < 2; ++mi) {
      v8f acc = {};
      for (int kt = 0; kt < 2; ++kt)
        acc = wmma16(load_a16(h1, 64, mi * 16, kt * 32, lane), load_b16p(W2p, 4, kt, ni, lane), acc);
      store_d_bnrelu(h2, 64, mi * 16, ni * 16, lane, acc, b2, g2, t2);
    }
  }
  __syncthreads();
  {  // layer2: (32x64)@(64x128) WMMA, NT=8
    for (int nn = 0; nn < 2; ++nn) {
      int ni = wave + nn * 4;
      for (int mi = 0; mi < 2; ++mi) {
        v8f acc = {};
        for (int kt = 0; kt < 2; ++kt)
          acc = wmma16(load_a16(h2, 64, mi * 16, kt * 32, lane), load_b16p(W3p, 8, kt, ni, lane), acc);
        store_d_bnrelu(h3, 128, mi * 16, ni * 16, lane, acc, b3, g3, t3);
      }
    }
  }
  __syncthreads();
  {  // max over 32 samples
    int c = tid;
    float m = -1e30f;
    for (int r = 0; r < 32; ++r) m = fmaxf(m, (float)h3[r * 128 + c]);
    outp[(size_t)bs * 128 + c] = m;
  }
}

// ---------------------------------------------------------------- SA2 fused: rows=64, 131->128->128->256, max
__global__ __launch_bounds__(256) void sa2_fused_kernel(
    const float* __restrict__ xyz1, const float* __restrict__ feat1,
    const float* __restrict__ cent, const int* __restrict__ gidx,
    const float* W1, const float* b1, const float* g1, const float* t1,
    const _Float16* W2p, const float* b2, const float* g2, const float* t2,
    const _Float16* W3p, const float* b3, const float* g3, const float* t3,
    float* __restrict__ outp) {
  __shared__ _Float16 smem[24832];   // in(64x132) | h1(64x128) ; h2 over in ; h3 over h1+
  __shared__ int sidx[64];
  const int bs = blockIdx.x;                // b*128 + s
  const int b = bs >> 7;
  const int tid = threadIdx.x, wave = tid >> 5, lane = tid & 31;
  _Float16* inb = smem;          // ld 132
  _Float16* h1  = smem + 8448;   // ld 128
  _Float16* h2  = smem;          // ld 128 (in is dead)
  _Float16* h3  = smem + 8448;   // ld 256 (h1 is dead)

  if (tid < 64) sidx[tid] = gidx[(size_t)bs * NS2 + tid];
  __syncthreads();
  const float cx = cent[(size_t)bs * 3 + 0], cy = cent[(size_t)bs * 3 + 1], cz = cent[(size_t)bs * 3 + 2];
  for (int i = tid; i < 64 * 131; i += 256) {
    int r = i / 131, ch = i - r * 131;
    int id = sidx[r];
    float v;
    if (ch < 3)
      v = xyz1[((size_t)b * S1 + id) * 3 + ch] - (ch == 0 ? cx : (ch == 1 ? cy : cz));
    else
      v = feat1[((size_t)b * S1 + id) * 128 + (ch - 3)];
    inb[r * 132 + ch] = (_Float16)v;
  }
  __syncthreads();
  {  // layer0: 131 -> 128 (scalar)
    int r = tid >> 2;
    int c0 = (tid & 3) * 32;
    for (int c = c0; c < c0 + 32; ++c) {
      float acc = b1[c];
      for (int k = 0; k < 131; ++k) acc += (float)inb[r * 132 + k] * W1[k * 128 + c];
      float y = g1[c] * acc + t1[c];
      h1[r * 128 + c] = (_Float16)fmaxf(y, 0.0f);
    }
  }
  __syncthreads();
  {  // layer1: (64x128)@(128x128) WMMA, NT=8
    int ni = wave;
    for (int mi = 0; mi < 4; ++mi) {
      v8f acc = {};
      for (int kt = 0; kt < 4; ++kt)
        acc = wmma16(load_a16(h1, 128, mi * 16, kt * 32, lane), load_b16p(W2p, 8, kt, ni, lane), acc);
      store_d_bnrelu(h2, 128, mi * 16, ni * 16, lane, acc, b2, g2, t2);
    }
  }
  __syncthreads();
  {  // layer2: (64x128)@(128x256) WMMA, NT=16
    for (int nn = 0; nn < 2; ++nn) {
      int ni = wave + nn * 8;
      for (int mi = 0; mi < 4; ++mi) {
        v8f acc = {};
        for (int kt = 0; kt < 4; ++kt)
          acc = wmma16(load_a16(h2, 128, mi * 16, kt * 32, lane), load_b16p(W3p, 16, kt, ni, lane), acc);
        store_d_bnrelu(h3, 256, mi * 16, ni * 16, lane, acc, b3, g3, t3);
      }
    }
  }
  __syncthreads();
  {  // max over 64 samples
    int c = tid;
    float m = -1e30f;
    for (int r = 0; r < 64; ++r) m = fmaxf(m, (float)h3[r * 256 + c]);
    outp[(size_t)bs * 256 + c] = m;
  }
}

// ---------------------------------------------------------------- SA3 (group_all), staged through global
__global__ __launch_bounds__(256) void sa3_layer1_kernel(
    const float* __restrict__ l2xyz, const float* __restrict__ l2p,
    const float* W1, const float* b1, const float* g1, const float* t1,
    _Float16* __restrict__ H) {     // 4096 x 256
  int row = blockIdx.x;             // b*128 + r
  __shared__ float sin[260];
  for (int i = threadIdx.x; i < 259; i += 256)
    sin[i] = (i < 3) ? l2xyz[(size_t)row * 3 + i] : l2p[(size_t)row * 256 + (i - 3)];
  __syncthreads();
  int c = threadIdx.x;
  float acc = b1[c];
  for (int k = 0; k < 259; ++k) acc += sin[k] * W1[k * 256 + c];
  float y = g1[c] * acc + t1[c];
  H[(size_t)row * 256 + c] = (_Float16)fmaxf(y, 0.0f);
}

// generic one-wave-per-16x16-tile WMMA GEMM with bn+relu (M,N mult of 16, K of 32), packed B
__global__ __launch_bounds__(32) void gemm_bn_relu_kernel(
    const _Float16* __restrict__ X, const _Float16* __restrict__ Wp,
    const float* bias, const float* g, const float* bt,
    _Float16* __restrict__ Y, int M, int K, int N) {
  int m0 = blockIdx.x * 16, nt = blockIdx.y;
  int lane = threadIdx.x;
  int NT = N >> 4;
  (void)M;
  v8f acc = {};
  int KT = K >> 5;
  for (int kt = 0; kt < KT; ++kt) {
    __builtin_prefetch(Wp + (((size_t)(kt + 1) * NT + nt) * 32 + lane) * 16, 0, 1);
    acc = wmma16(load_a16(X, K, m0, kt * 32, lane), load_b16p(Wp, NT, kt, nt, lane), acc);
  }
  store_d_bnrelu(Y, N, m0, nt * 16, lane, acc, bias, g, bt);
}

__global__ void sa3_max_kernel(const _Float16* __restrict__ H, float* __restrict__ feat) {
  int t = blockIdx.x * blockDim.x + threadIdx.x;     // BB*1024
  if (t >= BB * 1024) return;
  int b = t >> 10, c = t & 1023;
  float m = -1e30f;
  for (int r = 0; r < 128; ++r)
    m = fmaxf(m, (float)H[((size_t)(b * 128 + r)) * 1024 + c]);
  feat[t] = m;
}

// ---------------------------------------------------------------- FC head
__global__ __launch_bounds__(256) void fc_bn_relu_kernel(
    const float* __restrict__ X, const float* __restrict__ W,
    const float* bias, const float* g, const float* bt,
    float* __restrict__ Y, int K, int N) {
  int b = blockIdx.x;
  __shared__ float sx[1024];
  for (int i = threadIdx.x; i < K; i += 256) sx[i] = X[(size_t)b * K + i];
  __syncthreads();
  for (int c = threadIdx.x; c < N; c += 256) {
    float acc = bias[c];
    for (int k = 0; k < K; ++k) acc += sx[k] * W[(size_t)k * N + c];
    float y = g[c] * acc + bt[c];
    Y[(size_t)b * N + c] = fmaxf(y, 0.0f);
  }
}

__global__ __launch_bounds__(64) void fc3_logsoftmax_kernel(
    const float* __restrict__ X, const float* __restrict__ W, const float* __restrict__ bias,
    float* __restrict__ out, int K, int N) {
  int b = blockIdx.x;
  __shared__ float sx[256];
  __shared__ float logits[64];
  __shared__ float red[64];
  for (int i = threadIdx.x; i < K; i += 64) sx[i] = X[(size_t)b * K + i];
  __syncthreads();
  for (int c = threadIdx.x; c < N; c += 64) {
    float acc = bias[c];
    for (int k = 0; k < K; ++k) acc += sx[k] * W[(size_t)k * N + c];
    logits[c] = acc;
  }
  __syncthreads();
  float v = (threadIdx.x < N) ? logits[threadIdx.x] : -1e30f;
  red[threadIdx.x] = v; __syncthreads();
  for (int off = 32; off > 0; off >>= 1) {
    if (threadIdx.x < off) red[threadIdx.x] = fmaxf(red[threadIdx.x], red[threadIdx.x + off]);
    __syncthreads();
  }
  float mx = red[0]; __syncthreads();
  float e = (threadIdx.x < N) ? __expf(logits[threadIdx.x] - mx) : 0.0f;
  red[threadIdx.x] = e; __syncthreads();
  for (int off = 32; off > 0; off >>= 1) {
    if (threadIdx.x < off) red[threadIdx.x] += red[threadIdx.x + off];
    __syncthreads();
  }
  float lse = logf(red[0]) + mx;
  if (threadIdx.x < N) out[(size_t)b * N + threadIdx.x] = logits[threadIdx.x] - lse;
}

// ---------------------------------------------------------------- host
extern "C" void kernel_launch(void* const* d_in, const int* in_sizes, int n_in,
                              void* d_out, int out_size, void* d_ws, size_t ws_size,
                              hipStream_t stream) {
  (void)in_sizes; (void)n_in; (void)out_size; (void)ws_size;
  // input order: xyz | fc1(W,b,g,bt) fc2(..) fc3(W,b) | sa1 L0..L2 | sa2 L0..L2 | sa3 L0..L2
  const float* xyz = (const float*)d_in[0];
  const float* fc1W = (const float*)d_in[1];  const float* fc1b = (const float*)d_in[2];
  const float* fc1g = (const float*)d_in[3];  const float* fc1t = (const float*)d_in[4];
  const float* fc2W = (const float*)d_in[5];  const float* fc2b = (const float*)d_in[6];
  const float* fc2g = (const float*)d_in[7];  const float* fc2t = (const float*)d_in[8];
  const float* fc3W = (const float*)d_in[9];  const float* fc3b = (const float*)d_in[10];
  const float* s1W0 = (const float*)d_in[11]; const float* s1b0 = (const float*)d_in[12];
  const float* s1g0 = (const float*)d_in[13]; const float* s1t0 = (const float*)d_in[14];
  const float* s1W1 = (const float*)d_in[15]; const float* s1b1 = (const float*)d_in[16];
  const float* s1g1 = (const float*)d_in[17]; const float* s1t1 = (const float*)d_in[18];
  const float* s1W2 = (const float*)d_in[19]; const float* s1b2 = (const float*)d_in[20];
  const float* s1g2 = (const float*)d_in[21]; const float* s1t2 = (const float*)d_in[22];
  const float* s2W0 = (const float*)d_in[23]; const float* s2b0 = (const float*)d_in[24];
  const float* s2g0 = (const float*)d_in[25]; const float* s2t0 = (const float*)d_in[26];
  const float* s2W1 = (const float*)d_in[27]; const float* s2b1 = (const float*)d_in[28];
  const float* s2g1 = (const float*)d_in[29]; const float* s2t1 = (const float*)d_in[30];
  const float* s2W2 = (const float*)d_in[31]; const float* s2b2 = (const float*)d_in[32];
  const float* s2g2 = (const float*)d_in[33]; const float* s2t2 = (const float*)d_in[34];
  const float* s3W0 = (const float*)d_in[35]; const float* s3b0 = (const float*)d_in[36];
  const float* s3g0 = (const float*)d_in[37]; const float* s3t0 = (const float*)d_in[38];
  const float* s3W1 = (const float*)d_in[39]; const float* s3b1 = (const float*)d_in[40];
  const float* s3g1 = (const float*)d_in[41]; const float* s3t1 = (const float*)d_in[42];
  const float* s3W2 = (const float*)d_in[43]; const float* s3b2 = (const float*)d_in[44];
  const float* s3g2 = (const float*)d_in[45]; const float* s3t2 = (const float*)d_in[46];

  uint8_t* ws = (uint8_t*)d_ws;
  size_t off = 0;
  auto carve = [&](size_t bytes) -> void* {
    void* p = ws + off;
    off = (off + bytes + 255) & ~(size_t)255;
    return p;
  };
  float*    pts    = (float*)   carve((size_t)BB * NPT * 3 * 4);
  int*      fidx1  = (int*)     carve((size_t)BB * S1 * 4);
  float*    l1xyz  = (float*)   carve((size_t)BB * S1 * 3 * 4);
  int*      gidx1  = (int*)     carve((size_t)BB * S1 * NS1 * 4);
  float*    l1p    = (float*)   carve((size_t)BB * S1 * 128 * 4);
  int*      fidx2  = (int*)     carve((size_t)BB * S2 * 4);
  float*    l2xyz  = (float*)   carve((size_t)BB * S2 * 3 * 4);
  int*      gidx2  = (int*)     carve((size_t)BB * S2 * NS2 * 4);
  float*    l2p    = (float*)   carve((size_t)BB * S2 * 256 * 4);
  _Float16* H31    = (_Float16*)carve((size_t)4096 * 256 * 2);
  _Float16* H32    = (_Float16*)carve((size_t)4096 * 512 * 2);
  _Float16* H33    = (_Float16*)carve((size_t)4096 * 1024 * 2);
  float*    x1     = (float*)   carve((size_t)BB * 512 * 4);
  float*    x2     = (float*)   carve((size_t)BB * 256 * 4);
  _Float16* w2a    = (_Float16*)carve((size_t)64 * 64 * 2);
  _Float16* w3a    = (_Float16*)carve((size_t)64 * 128 * 2);
  _Float16* w2b    = (_Float16*)carve((size_t)128 * 128 * 2);
  _Float16* w3b    = (_Float16*)carve((size_t)128 * 256 * 2);
  _Float16* w2c    = (_Float16*)carve((size_t)256 * 512 * 2 + 32 * 16 * 2);   // +1 tile prefetch slack
  _Float16* w3c    = (_Float16*)carve((size_t)512 * 1024 * 2 + 32 * 16 * 2);

  float* logp = (float*)d_out;                 // 32 x 40
  float* feat = (float*)d_out + BB * 40;       // 32 x 1024

  // pack weights into WMMA B-fragment order (f16)
  pack_wmma_b_kernel<<<(64 * 64 + 255) / 256, 256, 0, stream>>>(s1W1, w2a, 64, 64);
  pack_wmma_b_kernel<<<(64 * 128 + 255) / 256, 256, 0, stream>>>(s1W2, w3a, 64, 128);
  pack_wmma_b_kernel<<<(128 * 128 + 255) / 256, 256, 0, stream>>>(s2W1, w2b, 128, 128);
  pack_wmma_b_kernel<<<(128 * 256 + 255) / 256, 256, 0, stream>>>(s2W2, w3b, 128, 256);
  pack_wmma_b_kernel<<<(256 * 512 + 255) / 256, 256, 0, stream>>>(s3W1, w2c, 256, 512);
  pack_wmma_b_kernel<<<(512 * 1024 + 255) / 256, 256, 0, stream>>>(s3W2, w3c, 512, 1024);

  // points (B,N,3)
  transpose_xyz_kernel<<<(BB * NPT * 3 + 255) / 256, 256, 0, stream>>>(xyz, pts);

  // ---- SA1
  fps_kernel<<<BB, 256, 0, stream>>>(pts, fidx1, NPT, S1);
  gather_xyz_kernel<<<(BB * S1 + 255) / 256, 256, 0, stream>>>(pts, fidx1, l1xyz, NPT, S1, BB * S1);
  ballquery_kernel<<<BB * S1, 32, 0, stream>>>(pts, l1xyz, gidx1, NPT, S1, NS1, 0.04f);
  sa1_fused_kernel<<<BB * S1, 128, 0, stream>>>(pts, l1xyz, gidx1,
      s1W0, s1b0, s1g0, s1t0, w2a, s1b1, s1g1, s1t1, w3a, s1b2, s1g2, s1t2, l1p);

  // ---- SA2
  fps_kernel<<<BB, 256, 0, stream>>>(l1xyz, fidx2, S1, S2);
  gather_xyz_kernel<<<(BB * S2 + 255) / 256, 256, 0, stream>>>(l1xyz, fidx2, l2xyz, S1, S2, BB * S2);
  ballquery_kernel<<<BB * S2, 32, 0, stream>>>(l1xyz, l2xyz, gidx2, S1, S2, NS2, 0.16f);
  sa2_fused_kernel<<<BB * S2, 256, 0, stream>>>(l1xyz, l1p, l2xyz, gidx2,
      s2W0, s2b0, s2g0, s2t0, w2b, s2b1, s2g1, s2t1, w3b, s2b2, s2g2, s2t2, l2p);

  // ---- SA3 (group_all)
  sa3_layer1_kernel<<<BB * S2, 256, 0, stream>>>(l2xyz, l2p, s3W0, s3b0, s3g0, s3t0, H31);
  gemm_bn_relu_kernel<<<dim3(4096 / 16, 512 / 16), 32, 0, stream>>>(
      H31, w2c, s3b1, s3g1, s3t1, H32, 4096, 256, 512);
  gemm_bn_relu_kernel<<<dim3(4096 / 16, 1024 / 16), 32, 0, stream>>>(
      H32, w3c, s3b2, s3g2, s3t2, H33, 4096, 512, 1024);
  sa3_max_kernel<<<(BB * 1024 + 255) / 256, 256, 0, stream>>>(H33, feat);

  // ---- head
  fc_bn_relu_kernel<<<BB, 256, 0, stream>>>(feat, fc1W, fc1b, fc1g, fc1t, x1, 1024, 512);
  fc_bn_relu_kernel<<<BB, 256, 0, stream>>>(x1, fc2W, fc2b, fc2g, fc2t, x2, 512, 256);
  fc3_logsoftmax_kernel<<<BB, 64, 0, stream>>>(x2, fc3W, fc3b, logp, 256, 40);
}
